// Encoder_35416300323553
// MI455X (gfx1250) — matrix operand order repaired and checked by reading, not compiled, not run on hardware
//
#include <hip/hip_runtime.h>
#include <hip/hip_bf16.h>
#include <math.h>

#define L_   6
#define H_   8
#define D_   512
#define DK_  64
#define DFF_ 2048
#define B_   8
#define S_   1024
#define EPS_ 1e-5f

typedef __attribute__((ext_vector_type(16))) __bf16 v16bf;
typedef __attribute__((ext_vector_type(8)))  float  v8f;

static __device__ __forceinline__ unsigned short f2bf(float f) {
  union { float f; unsigned u; } c; c.f = f;
  return (unsigned short)((c.u + 0x7fffu + ((c.u >> 16) & 1u)) >> 16); // RNE
}
static __device__ __forceinline__ unsigned pk2(float a, float b) {
  return (unsigned)f2bf(a) | ((unsigned)f2bf(b) << 16);
}

union FragBF { v16bf v; uint4 q[2]; unsigned short h[16]; };

static __device__ __forceinline__ v8f wmma_bf16(const FragBF& a, const FragBF& b, v8f c) {
  return __builtin_amdgcn_wmma_f32_16x16x32_bf16(false, a.v, false, b.v, (short)0, c,
                                                 false, false);
}

// ---------------------------------------------------------------------------
// bf16-WMMA GEMM: C[M][N] = A[M][K] * B.  Block tile 128x128, 8 waves (4x2),
// wave tile 32x64 (8 WMMA / 12 ds_load_b128 per k-step), double-buffered LDS,
// prefetch 2 tiles ahead.
// B column n lives at: Bm + (n>>6)*bstride + (n&63) + k*ldb
//   (bstride==64  -> plain row-major B;  bstride==D*DK -> per-head QKV blocks)
// ---------------------------------------------------------------------------
template<int HAS_BIAS, int HAS_RESID, int RELU>
__launch_bounds__(256, 2)
__global__ void gemm_bf16_kernel(const float* __restrict__ A,
                                 const float* __restrict__ Bm,
                                 const float* __restrict__ bias,
                                 const float* __restrict__ resid,
                                 float* __restrict__ C,
                                 int K, int lda, int ldb, long bstride, int ldc)
{
  __shared__ __align__(16) unsigned short Asm[2][128 * 40]; // [row][k] pitch 40
  __shared__ __align__(16) unsigned short Bsm[2][128 * 40]; // [n][k]   pitch 40

  const int  t    = threadIdx.x;
  const int  lane = t & 31;
  const int  wave = t >> 5;
  const int  wm   = wave & 3;     // 32-row slice
  const int  wn   = wave >> 2;    // 64-col slice
  const long m0   = (long)blockIdx.y * 128;
  const long n0   = (long)blockIdx.x * 128;

  // A staging: thread -> (row, 16-wide k half)
  const int arow = t >> 1;
  const int acb  = (t & 1) * 16;
  // B staging: thread -> (column, 16-wide k chunk); conflict-free LDS stores
  const int bn   = t & 127;
  const int bkb  = (t >> 7) * 16;

  const float* Ab = A + (m0 + arow) * (long)lda + acb;
  const long  bc  = n0 + bn;
  const float* Bb = Bm + (bc >> 6) * bstride + (bc & 63) + (long)bkb * ldb;

  const int row16 = lane & 15;
  const int hi    = lane >> 4;
  const int kloA  = hi * 8;
  const int kloB  = hi * 8;

  v8f acc[2][4] = {};

  const int nsteps = K >> 5;

  float4 ra0, ra1, ra2, ra3;
  float  rb[16];

  auto gload = [&](int it) {
    const float* s = Ab + it * 32;
    ra0 = *(const float4*)(s);
    ra1 = *(const float4*)(s + 4);
    ra2 = *(const float4*)(s + 8);
    ra3 = *(const float4*)(s + 12);
    const float* sb = Bb + (long)it * 32 * ldb;
#pragma unroll
    for (int i = 0; i < 16; ++i) rb[i] = sb[(long)i * ldb];
  };
  auto stage = [&](int buf) {
    uint2* d = (uint2*)(Asm[buf] + arow * 40 + acb);
    d[0] = make_uint2(pk2(ra0.x, ra0.y), pk2(ra0.z, ra0.w));
    d[1] = make_uint2(pk2(ra1.x, ra1.y), pk2(ra1.z, ra1.w));
    d[2] = make_uint2(pk2(ra2.x, ra2.y), pk2(ra2.z, ra2.w));
    d[3] = make_uint2(pk2(ra3.x, ra3.y), pk2(ra3.z, ra3.w));
    uint2* db = (uint2*)(Bsm[buf] + bn * 40 + bkb);
    db[0] = make_uint2(pk2(rb[0],  rb[1]),  pk2(rb[2],  rb[3]));
    db[1] = make_uint2(pk2(rb[4],  rb[5]),  pk2(rb[6],  rb[7]));
    db[2] = make_uint2(pk2(rb[8],  rb[9]),  pk2(rb[10], rb[11]));
    db[3] = make_uint2(pk2(rb[12], rb[13]), pk2(rb[14], rb[15]));
  };

  gload(0);
  stage(0);
  __syncthreads();

  for (int it = 0; it < nsteps; ++it) {
    const int cur = it & 1;
    if (it + 1 < nsteps) gload(it + 1);
    if (it + 2 < nsteps) {
      __builtin_prefetch(Ab + (it + 2) * 32, 0, 1);
      __builtin_prefetch(Bb + (long)(it + 2) * 32 * ldb, 0, 1);
    }

    FragBF a0, a1, b[4];
    {
      const unsigned short* As = Asm[cur];
      const unsigned short* Bs = Bsm[cur];
      const unsigned short* p  = As + (wm * 32 + row16) * 40 + kloA;
      a0.q[0] = *(const uint4*)(p);        a0.q[1] = *(const uint4*)(p + 16);
      const unsigned short* p1 = As + (wm * 32 + 16 + row16) * 40 + kloA;
      a1.q[0] = *(const uint4*)(p1);       a1.q[1] = *(const uint4*)(p1 + 16);
#pragma unroll
      for (int j = 0; j < 4; ++j) {
        const unsigned short* pb = Bs + (wn * 64 + j * 16 + row16) * 40 + kloB;
        b[j].q[0] = *(const uint4*)(pb);
        b[j].q[1] = *(const uint4*)(pb + 16);
      }
    }
#pragma unroll
    for (int j = 0; j < 4; ++j) {
      acc[0][j] = wmma_bf16(a0, b[j], acc[0][j]);
      acc[1][j] = wmma_bf16(a1, b[j], acc[1][j]);
    }

    if (it + 1 < nsteps) stage(cur ^ 1);
    __syncthreads();
  }

#pragma unroll
  for (int i = 0; i < 2; ++i) {
#pragma unroll
    for (int j = 0; j < 4; ++j) {
      const long mg = m0 + wm * 32 + i * 16 + hi * 8;
      const long ng = n0 + wn * 64 + j * 16 + row16;
      float* cp = C + mg * ldc + ng;
      const float bv = HAS_BIAS ? bias[ng] : 0.f;
      const float* rp = HAS_RESID ? (resid + mg * ldc + ng) : nullptr;
#pragma unroll
      for (int e = 0; e < 8; ++e) {
        float vv = acc[i][j][e] + bv;
        if (HAS_RESID) vv += rp[(long)e * ldc];
        if (RELU) vv = fmaxf(vv, 0.f);
        cp[(long)e * ldc] = vv;
      }
    }
  }
}

// ---------------------------------------------------------------------------
// Flash attention: q/k/v in [B][S][H*DK].  Block = 4 waves, each wave owns 16
// query rows.  64-key tiles staged to LDS; online softmax; WMMA throughout.
// ---------------------------------------------------------------------------
__launch_bounds__(128, 2)
__global__ void attn_kernel(const float* __restrict__ q,
                            const float* __restrict__ k,
                            const float* __restrict__ v,
                            float* __restrict__ ctx)
{
  __shared__ __align__(16) unsigned short Kt[64 * 72];    // [key][feat]
  __shared__ __align__(16) unsigned short Vt[64 * 72];    // [feat][key]
  __shared__ __align__(16) unsigned short Pt[4][16 * 72]; // per-wave [row][key]

  const int t     = threadIdx.x;
  const int lane  = t & 31;
  const int wave  = t >> 5;
  const int b     = blockIdx.y / H_;
  const int h     = blockIdx.y % H_;
  const int q0    = blockIdx.x * 64 + wave * 16;
  const int row16 = lane & 15;
  const int hi    = lane >> 4;

  const long base = (long)b * S_ * D_ + h * DK_;
  const float scale = 0.125f; // 1/sqrt(64)

  // Q fragments (A layout), features 0..31 / 32..63
  FragBF aq0, aq1;
  {
    const float* qr = q + base + (long)(q0 + row16) * D_;
    const int klo = hi * 16;
#pragma unroll
    for (int i = 0; i < 8; ++i) {
      aq0.h[i]     = f2bf(qr[klo + i]);
      aq0.h[8 + i] = f2bf(qr[klo + 8 + i]);
      aq1.h[i]     = f2bf(qr[32 + klo + i]);
      aq1.h[8 + i] = f2bf(qr[32 + klo + 8 + i]);
    }
  }

  float m_run[8], l_run[8];
  v8f o0 = {}, o1 = {}, o2 = {}, o3 = {};
#pragma unroll
  for (int e = 0; e < 8; ++e) { m_run[e] = -INFINITY; l_run[e] = 0.f; }

  const int kkey = t >> 1;         // K staging: key 0..63
  const int kfb  = (t & 1) * 32;   //            feat half
  const int vk4  = (t & 15) * 4;   // V staging: key group of 4
  const int vfb  = (t >> 4) * 8;   //            feat group of 8
  const int kloB = hi * 16;
  unsigned short* Pw = Pt[wave];

  for (int t0 = 0; t0 < S_; t0 += 64) {
    __syncthreads();
    { // stage K tile [64 key][64 feat] bf16, row-major
      const float* s = k + base + (long)(t0 + kkey) * D_ + kfb;
      uint2* d = (uint2*)(Kt + kkey * 72 + kfb);
#pragma unroll
      for (int m = 0; m < 4; ++m) {
        float4 f0 = *(const float4*)(s + 8 * m);
        float4 f1 = *(const float4*)(s + 8 * m + 4);
        d[2 * m]     = make_uint2(pk2(f0.x, f0.y), pk2(f0.z, f0.w));
        d[2 * m + 1] = make_uint2(pk2(f1.x, f1.y), pk2(f1.z, f1.w));
      }
    }
    { // stage V tile transposed -> [64 feat][64 key], packed uint2 stores
      float fv[4][8];
#pragma unroll
      for (int kk = 0; kk < 4; ++kk) {
        const float* s = v + base + (long)(t0 + vk4 + kk) * D_ + vfb;
        float4 f0 = *(const float4*)(s);
        float4 f1 = *(const float4*)(s + 4);
        fv[kk][0] = f0.x; fv[kk][1] = f0.y; fv[kk][2] = f0.z; fv[kk][3] = f0.w;
        fv[kk][4] = f1.x; fv[kk][5] = f1.y; fv[kk][6] = f1.z; fv[kk][7] = f1.w;
      }
#pragma unroll
      for (int i = 0; i < 8; ++i) {
        *(uint2*)(Vt + (vfb + i) * 72 + vk4) =
            make_uint2(pk2(fv[0][i], fv[1][i]), pk2(fv[2][i], fv[3][i]));
      }
    }
    if (t0 + 64 < S_) {
      __builtin_prefetch(k + base + (long)(t0 + 64 + kkey) * D_ + kfb, 0, 1);
      __builtin_prefetch(v + base + (long)(t0 + 64 + vk4) * D_ + vfb, 0, 1);
    }
    __syncthreads();

    // scores: 4 sub-tiles of 16 keys
    v8f s0 = {}, s1 = {}, s2 = {}, s3 = {};
#define SCORE(ss, dst) { \
      const unsigned short* p0 = Kt + ((ss) * 16 + row16) * 72 + kloB; \
      FragBF bf0, bf1; \
      bf0.q[0] = *(const uint4*)(p0);      bf0.q[1] = *(const uint4*)(p0 + 8); \
      bf1.q[0] = *(const uint4*)(p0 + 32); bf1.q[1] = *(const uint4*)(p0 + 40); \
      dst = wmma_bf16(aq0, bf0, dst); dst = wmma_bf16(aq1, bf1, dst); }
    SCORE(0, s0) SCORE(1, s1) SCORE(2, s2) SCORE(3, s3)
#undef SCORE

    // online softmax over the 64-key tile
#pragma unroll
    for (int e = 0; e < 8; ++e) {
      float x0 = s0[e] * scale, x1 = s1[e] * scale;
      float x2 = s2[e] * scale, x3 = s3[e] * scale;
      float mx = fmaxf(fmaxf(x0, x1), fmaxf(x2, x3));
      mx = fmaxf(mx, __shfl_xor(mx, 1, 32));
      mx = fmaxf(mx, __shfl_xor(mx, 2, 32));
      mx = fmaxf(mx, __shfl_xor(mx, 4, 32));
      mx = fmaxf(mx, __shfl_xor(mx, 8, 32));
      float mnew  = fmaxf(m_run[e], mx);
      float alpha = __expf(m_run[e] - mnew);
      float p0 = __expf(x0 - mnew), p1 = __expf(x1 - mnew);
      float p2 = __expf(x2 - mnew), p3 = __expf(x3 - mnew);
      float ps = (p0 + p1) + (p2 + p3);
      ps += __shfl_xor(ps, 1, 32);
      ps += __shfl_xor(ps, 2, 32);
      ps += __shfl_xor(ps, 4, 32);
      ps += __shfl_xor(ps, 8, 32);
      l_run[e] = l_run[e] * alpha + ps;
      m_run[e] = mnew;
      o0[e] *= alpha; o1[e] *= alpha; o2[e] *= alpha; o3[e] *= alpha;
      unsigned short* pr = Pw + (e + hi * 8) * 72 + row16;
      pr[0]  = f2bf(p0);
      pr[16] = f2bf(p1);
      pr[32] = f2bf(p2);
      pr[48] = f2bf(p3);
    }

    // P x V: two K=32 steps over the 64 keys
#pragma unroll
    for (int ks = 0; ks < 2; ++ks) {
      FragBF pf;
      const unsigned short* pp = Pw + row16 * 72 + ks * 32 + hi * 16;
      pf.q[0] = *(const uint4*)(pp);
      pf.q[1] = *(const uint4*)(pp + 8);
      const int off = ks * 32 + kloB;
      FragBF bv0, bv1, bv2, bv3;
      const unsigned short* v0p = Vt + (0 * 16 + row16) * 72 + off;
      bv0.q[0] = *(const uint4*)(v0p); bv0.q[1] = *(const uint4*)(v0p + 8);
      const unsigned short* v1p = Vt + (1 * 16 + row16) * 72 + off;
      bv1.q[0] = *(const uint4*)(v1p); bv1.q[1] = *(const uint4*)(v1p + 8);
      const unsigned short* v2p = Vt + (2 * 16 + row16) * 72 + off;
      bv2.q[0] = *(const uint4*)(v2p); bv2.q[1] = *(const uint4*)(v2p + 8);
      const unsigned short* v3p = Vt + (3 * 16 + row16) * 72 + off;
      bv3.q[0] = *(const uint4*)(v3p); bv3.q[1] = *(const uint4*)(v3p + 8);
      o0 = wmma_bf16(pf, bv0, o0);
      o1 = wmma_bf16(pf, bv1, o1);
      o2 = wmma_bf16(pf, bv2, o2);
      o3 = wmma_bf16(pf, bv3, o3);
    }
  }

  // normalize, write ctx [B][S][H*DK]
#pragma unroll
  for (int e = 0; e < 8; ++e) {
    const float inv = 1.f / l_run[e];
    float* dst = ctx + base + (long)(q0 + e + hi * 8) * D_;
    dst[row16]      = o0[e] * inv;
    dst[16 + row16] = o1[e] * inv;
    dst[32 + row16] = o2[e] * inv;
    dst[48 + row16] = o3[e] * inv;
  }
}

// ---------------------------------------------------------------------------
// LayerNorm over D=512, one row per block (256 threads, 2 elems/thread)
// ---------------------------------------------------------------------------
__launch_bounds__(256, 4)
__global__ void ln_kernel(const float* __restrict__ x,
                          const float* __restrict__ g,
                          const float* __restrict__ be,
                          float* __restrict__ y)
{
  __shared__ float red[8];
  const int  t   = threadIdx.x;
  const long row = blockIdx.x;
  const float* xr = x + row * D_;
  float a = xr[t], b = xr[t + 256];

  float s = a + b;
#pragma unroll
  for (int m = 1; m < 32; m <<= 1) s += __shfl_xor(s, m, 32);
  if ((t & 31) == 0) red[t >> 5] = s;
  __syncthreads();
  float mean = (red[0] + red[1] + red[2] + red[3] +
                red[4] + red[5] + red[6] + red[7]) * (1.f / 512.f);

  float da = a - mean, db = b - mean;
  float vs = da * da + db * db;
#pragma unroll
  for (int m = 1; m < 32; m <<= 1) vs += __shfl_xor(vs, m, 32);
  __syncthreads();
  if ((t & 31) == 0) red[t >> 5] = vs;
  __syncthreads();
  float var = (red[0] + red[1] + red[2] + red[3] +
               red[4] + red[5] + red[6] + red[7]) * (1.f / 512.f);
  float rstd = rsqrtf(var + EPS_);

  y[row * D_ + t]       = da * rstd * g[t]       + be[t];
  y[row * D_ + t + 256] = db * rstd * g[t + 256] + be[t + 256];
}

// ---------------------------------------------------------------------------
extern "C" void kernel_launch(void* const* d_in, const int* in_sizes, int n_in,
                              void* d_out, int out_size, void* d_ws, size_t ws_size,
                              hipStream_t stream)
{
  (void)in_sizes; (void)n_in; (void)out_size; (void)ws_size;
  const float* x   = (const float*)d_in[0];
  const float* Wq  = (const float*)d_in[1];
  const float* Wk  = (const float*)d_in[2];
  const float* Wv  = (const float*)d_in[3];
  const float* Wo  = (const float*)d_in[4];
  const float* g1  = (const float*)d_in[5];
  const float* be1 = (const float*)d_in[6];
  const float* W1  = (const float*)d_in[7];
  const float* b1  = (const float*)d_in[8];
  const float* W2  = (const float*)d_in[9];
  const float* b2  = (const float*)d_in[10];
  const float* g2  = (const float*)d_in[11];
  const float* be2 = (const float*)d_in[12];
  float* out = (float*)d_out;

  float* ws = (float*)d_ws;
  const size_t MT = (size_t)B_ * S_;      // 8192 tokens
  float* qb = ws;                         // MT*512
  float* kb = qb + MT * D_;
  float* vb = kb + MT * D_;
  float* cx = vb + MT * D_;               // ctx, then mha
  float* t1 = cx + MT * D_;
  float* hb = t1 + MT * D_;
  float* ff = hb + MT * D_;               // MT*2048

  const dim3 g512(4, 64), g2048(16, 64);  // N/128 x M/128
  const float* cur = x;

  for (int l = 0; l < L_; ++l) {
    const float* Wq_l = Wq + (size_t)l * H_ * D_ * DK_;
    const float* Wk_l = Wk + (size_t)l * H_ * D_ * DK_;
    const float* Wv_l = Wv + (size_t)l * H_ * D_ * DK_;
    const float* Wo_l = Wo + (size_t)l * D_ * D_;
    const float* W1_l = W1 + (size_t)l * D_ * DFF_;
    const float* W2_l = W2 + (size_t)l * DFF_ * D_;

    // QKV projections (per-head weight blocks, bstride = D*DK)
    gemm_bf16_kernel<0, 0, 0><<<g512, 256, 0, stream>>>(
        cur, Wq_l, nullptr, nullptr, qb, D_, D_, DK_, (long)D_ * DK_, D_);
    gemm_bf16_kernel<0, 0, 0><<<g512, 256, 0, stream>>>(
        cur, Wk_l, nullptr, nullptr, kb, D_, D_, DK_, (long)D_ * DK_, D_);
    gemm_bf16_kernel<0, 0, 0><<<g512, 256, 0, stream>>>(
        cur, Wv_l, nullptr, nullptr, vb, D_, D_, DK_, (long)D_ * DK_, D_);
    // attention
    attn_kernel<<<dim3(S_ / 64, B_ * H_), 128, 0, stream>>>(qb, kb, vb, cx);
    // output projection + residual, then LN -> mha (reuse cx)
    gemm_bf16_kernel<0, 1, 0><<<g512, 256, 0, stream>>>(
        cx, Wo_l, nullptr, cur, t1, D_, D_, D_, 64L, D_);
    ln_kernel<<<MT, 256, 0, stream>>>(t1, g1 + l * D_, be1 + l * D_, cx);
    // FFN
    gemm_bf16_kernel<1, 0, 1><<<g2048, 256, 0, stream>>>(
        cx, W1_l, b1 + (size_t)l * DFF_, nullptr, ff, D_, D_, DFF_, 64L, DFF_);
    gemm_bf16_kernel<1, 1, 0><<<g512, 256, 0, stream>>>(
        ff, W2_l, b2 + (size_t)l * D_, cx, t1, DFF_, DFF_, D_, 64L, D_);
    ln_kernel<<<MT, 256, 0, stream>>>(t1, g2 + l * D_, be2 + l * D_,
                                      (l == L_ - 1) ? out : hb);
    cur = hb;
  }
}
